// Network_72524817760271
// MI455X (gfx1250) — matrix operand order repaired and checked by reading, not compiled, not run on hardware
//
#include <hip/hip_runtime.h>
#include <hip/hip_bf16.h>
#include <stdint.h>

// ---------- types ----------
typedef _Float16 h16;
typedef __attribute__((ext_vector_type(8)))  _Float16 v8h;
typedef __attribute__((ext_vector_type(16))) _Float16 v16h;
typedef __attribute__((ext_vector_type(8)))  float    v8f;

union AFrag { v16h v; v8h h[2]; };

#define NLVL 5
#define TOTPIX 51150
#define PADPIX 52720   // sum of (H+2)*(W+2)

// ---------- zero-fill f16 buffer (halo init) ----------
__global__ void zero_f16(h16* __restrict__ p, int n) {
    int i = blockIdx.x * blockDim.x + threadIdx.x;
    if (i < n) p[i] = (h16)0.f;
}

// ---------- NCHW f32 -> padded NHWC f16 (interior only; halo pre-zeroed) ----------
__global__ void to_nhwc_f16(const float* __restrict__ in, h16* __restrict__ out,
                            int H, int W) {
    int t = blockIdx.x * blockDim.x + threadIdx.x;
    int HW = H * W;
    if (t < HW * 256) {
        int p = t >> 8, c = t & 255;
        int y = p / W, x = p % W;
        out[((size_t)((y + 1) * (W + 2) + (x + 1)) << 8) + c] = (h16)in[(size_t)c * HW + p];
    }
}

// ---------- pack tower weights (OIHW f32 -> WMMA-B fragment order f16) ----------
// packed index: ((kc*16 + n16)*32 + lane)*16 + e
//   K = kc*32 + (e>>3)*16 + (lane>>4)*8 + (e&7)   (K = tap*256 + ci, tap-major)
//   N = n16*16 + (lane&15)
__global__ void pack_tower_w(const float* __restrict__ w, h16* __restrict__ dst, int layer) {
    int idx = blockIdx.x * blockDim.x + threadIdx.x;
    if (idx >= 72 * 16 * 32 * 16) return;
    int e    = idx & 15;
    int lane = (idx >> 4) & 31;
    int n16  = (idx >> 9) & 15;
    int kc   = idx >> 13;
    int K  = kc * 32 + (e >> 3) * 16 + (lane >> 4) * 8 + (e & 7);
    int N  = n16 * 16 + (lane & 15);
    int tap = K >> 8, ci = K & 255;
    int ky = tap / 3, kx = tap % 3;
    float v = w[((((size_t)layer * 256 + N) * 256 + ci) * 3 + ky) * 3 + kx];
    dst[idx] = (h16)v;
}

// mode 0: col0=score_w ; mode 1: col0..7=pred_w, col8=ctn_w
__global__ void pack_head_w(const float* __restrict__ w0, const float* __restrict__ w1,
                            h16* __restrict__ dst, int mode) {
    int idx = blockIdx.x * blockDim.x + threadIdx.x;
    if (idx >= 72 * 32 * 16) return;
    int e    = idx & 15;
    int lane = (idx >> 4) & 31;
    int kc   = idx >> 9;
    int K  = kc * 32 + (e >> 3) * 16 + (lane >> 4) * 8 + (e & 7);
    int n  = lane & 15;
    int tap = K >> 8, ci = K & 255;
    int ky = tap / 3, kx = tap % 3;
    float v = 0.f;
    if (mode == 0) {
        if (n == 0) v = w0[(((size_t)ci) * 3 + ky) * 3 + kx];
    } else {
        if (n < 8)       v = w0[((((size_t)n * 256) + ci) * 3 + ky) * 3 + kx];
        else if (n == 8) v = w1[(((size_t)ci) * 3 + ky) * 3 + kx];
    }
    dst[idx] = (h16)v;
}

__global__ void build_head_bias(const float* sb, const float* pb, const float* cb,
                                float* bA, float* bB) {
    int t = threadIdx.x;
    if (t < 16) {
        bA[t] = (t == 0) ? sb[0] : 0.f;
        bB[t] = (t < 8) ? pb[t] : ((t == 8) ? cb[0] : 0.f);
    }
}

// ---------- implicit-GEMM conv3x3 via v_wmma_f32_16x16x32_f16 ----------
// act: padded NHWC f16 [(H+2)*(W+2)][256], zero halo => NO bounds checks in K loop.
// GEMM: M=H*W output pixels, N=NOUT, K=2304 (tap-major).
// NOUT=256 (tower, ReLU, padded f16 out): 4 waves 2x2 -> 64x64 tile.
// NOUT=16  (heads, unpadded f32 out):     4 waves 4x1 -> 128x16 tile.
template <int NOUT, bool TOWER>
__global__ __launch_bounds__(128) void conv3x3_wmma(
    const h16* __restrict__ act, const h16* __restrict__ wpk,
    const float* __restrict__ bias, h16* __restrict__ outH,
    float* __restrict__ outF, int H, int W) {

    constexpr int NF = (NOUT == 256) ? 2 : 1;
    constexpr int CSTRIDE = (NOUT / 16) * 512;   // packed-B halfs per K-chunk
    const int HW = H * W;
    const int Wp = W + 2;
    const int lane = threadIdx.x & 31;
    const int wave = threadIdx.x >> 5;

    int waveM, waveN;
    if (NOUT == 256) { waveM = (wave & 1) * 32; waveN = (wave >> 1) * 32; }
    else             { waveM = wave * 32;       waveN = 0; }
    const int Mbase = blockIdx.x * ((NOUT == 256) ? 64 : 128) + waveM;
    const int Nbase = blockIdx.y * 64 + waveN;

    const int kbaseA = (lane >> 4) * 8;          // ISA 16-bit A-layout lane K base

    // per-lane padded input-row base pointers (clamped rows load harmlessly)
    const h16* abase[2];
    #pragma unroll
    for (int mf = 0; mf < 2; ++mf) {
        int m = Mbase + mf * 16 + (lane & 15);
        int mm = (m < HW) ? m : 0;
        int y = mm / W, x = mm % W;
        abase[mf] = act + ((size_t)((y + 1) * Wp + (x + 1)) << 8) + kbaseA;
    }
    const h16* wl = wpk + (size_t)(Nbase >> 4) * 512 + (size_t)lane * 16;

    v8f zf = {};
    v8f acc[2][NF];
    #pragma unroll
    for (int mf = 0; mf < 2; ++mf)
        #pragma unroll
        for (int nf = 0; nf < NF; ++nf) acc[mf][nf] = zf;

    for (int tap = 0; tap < 9; ++tap) {
        const int dy = tap / 3 - 1;
        const int dx = tap % 3 - 1;
        const int tofs = (dy * Wp + dx) << 8;    // pixel offset in halfs
        const h16* ap0 = abase[0] + tofs;
        const h16* ap1 = abase[1] + tofs;
        const h16* wt  = wl + (size_t)tap * 8 * CSTRIDE;

        #pragma unroll 4
        for (int cc = 0; cc < 8; ++cc) {         // 8 channel chunks of 32
            const int ci0 = cc * 32;
            AFrag a0, a1;
            a0.h[0] = *(const v8h*)(ap0 + ci0);
            a0.h[1] = *(const v8h*)(ap0 + ci0 + 16);
            a1.h[0] = *(const v8h*)(ap1 + ci0);
            a1.h[1] = *(const v8h*)(ap1 + ci0 + 16);

            v16h b0 = *(const v16h*)(wt + cc * CSTRIDE);
            acc[0][0] = __builtin_amdgcn_wmma_f32_16x16x32_f16(
                false, a0.v, false, b0, (short)0, acc[0][0], false, false);
            acc[1][0] = __builtin_amdgcn_wmma_f32_16x16x32_f16(
                false, a1.v, false, b0, (short)0, acc[1][0], false, false);
            if constexpr (NF == 2) {
                v16h b1 = *(const v16h*)(wt + cc * CSTRIDE + 512);
                acc[0][1] = __builtin_amdgcn_wmma_f32_16x16x32_f16(
                    false, a0.v, false, b1, (short)0, acc[0][1], false, false);
                acc[1][1] = __builtin_amdgcn_wmma_f32_16x16x32_f16(
                    false, a1.v, false, b1, (short)0, acc[1][1], false, false);
            }
        }
    }

    // epilogue: ISA C/D layout: col = lane&15, row = v + 8*(lane>>4)
    const int nl = lane & 15;
    const int mh = 8 * (lane >> 4);
    #pragma unroll
    for (int mf = 0; mf < 2; ++mf) {
        #pragma unroll
        for (int nf = 0; nf < NF; ++nf) {
            const int n = Nbase + nf * 16 + nl;
            #pragma unroll
            for (int v = 0; v < 8; ++v) {
                int m = Mbase + mf * 16 + mh + v;
                if (m < HW) {
                    float val = acc[mf][nf][v] + bias[n];
                    if constexpr (TOWER) {
                        int y = m / W, x = m % W;
                        outH[((size_t)((y + 1) * Wp + (x + 1)) << 8) + n] =
                            (h16)fmaxf(val, 0.f);
                    } else {
                        outF[(size_t)m * 16 + n] = val;
                    }
                }
            }
        }
    }
}

// ---------- sigmoid(cls)*sigmoid(ctn), gather reg deltas ----------
__global__ void fuse_scores(const float* __restrict__ hA, const float* __restrict__ hB,
                            float* __restrict__ sc, float* __restrict__ rg, int n) {
    int p = blockIdx.x * blockDim.x + threadIdx.x;
    if (p < n) {
        float c = hA[(size_t)p * 16 + 0];
        float t = hB[(size_t)p * 16 + 8];
        sc[p] = (1.f / (1.f + __expf(-c))) * (1.f / (1.f + __expf(-t)));
        rg[(size_t)p * 4 + 0] = hB[(size_t)p * 16 + 0];
        rg[(size_t)p * 4 + 1] = hB[(size_t)p * 16 + 1];
        rg[(size_t)p * 4 + 2] = hB[(size_t)p * 16 + 2];
        rg[(size_t)p * 4 + 3] = hB[(size_t)p * 16 + 3];
    }
}

// ---------- bbox decode ----------
__device__ __forceinline__ void decode_write(float* out, int row, int i, int W,
                                             float stride, float score,
                                             const float* __restrict__ rg) {
    int y = i / W, x = i % W;
    float cx0 = (x + 0.5f) * stride, cy0 = (y + 0.5f) * stride;
    float hf = 4.0f * stride;                         // 0.5 * ANCHOR_SIZE * stride
    float x1 = cx0 - hf, y1 = cy0 - hf, x2 = cx0 + hf, y2 = cy0 + hf;
    float w = x2 - x1 + 1.f, h = y2 - y1 + 1.f;
    float cx = x1 + 0.5f * w, cy = y1 + 0.5f * h;
    float pcx = cx + rg[0] * w, pcy = cy + rg[1] * h;
    const float maxd = 4.135166556742356f;            // log(1000/16)
    float pw = w * __expf(fminf(rg[2], maxd));
    float ph = h * __expf(fminf(rg[3], maxd));
    out[(size_t)row * 6 + 0] = pcx - 0.5f * pw;
    out[(size_t)row * 6 + 1] = pcy - 0.5f * ph;
    out[(size_t)row * 6 + 2] = pcx + 0.5f * pw - 1.f;
    out[(size_t)row * 6 + 3] = pcy + 0.5f * ph - 1.f;
    out[(size_t)row * 6 + 4] = score;
    out[(size_t)row * 6 + 5] = 1.0f;                  // class tag
}

__global__ void emit_small(const float* __restrict__ sc, const float* __restrict__ rg,
                           float* __restrict__ out, int pixOfs, int rowOfs,
                           int n, int W, float stride) {
    int i = blockIdx.x * blockDim.x + threadIdx.x;
    if (i < n)
        decode_write(out, rowOfs + i, i, W, stride, sc[pixOfs + i], rg + (size_t)(pixOfs + i) * 4);
}

// ---------- deterministic block scan ----------
__device__ int block_exscan(int v, int t, int* buf, int* tot) {
    buf[t] = v; __syncthreads();
    for (int d = 1; d < 1024; d <<= 1) {
        int x = (t >= d) ? buf[t - d] : 0;
        __syncthreads();
        buf[t] += x;
        __syncthreads();
    }
    int incl = buf[t];
    int total = buf[1023];
    __syncthreads();
    *tot = total;
    return incl - v;
}

// ---------- per-level top-1000 (histogram select) + sort + decode ----------
__global__ __launch_bounds__(1024) void topk_emit(const float* __restrict__ scores,
                                                  const float* __restrict__ rg,
                                                  float* __restrict__ out) {
    const int pixOfsA[3] = {750, 3150, 12750};
    const int nA[3]      = {2400, 9600, 38400};
    const int wA[3]      = {60, 120, 240};
    const float strA[3]  = {32.f, 16.f, 8.f};
    const int rowOfsA[3] = {750, 1750, 2750};

    __shared__ int   hist[4096];
    __shared__ float selS[1024];
    __shared__ int   selI[1024];
    __shared__ int   scanbuf[1024];
    __shared__ int   sh_thresh, sh_needEq, sh_runHi, sh_runEq;

    const int lvl = blockIdx.x;
    const int t   = threadIdx.x;
    const int n   = nA[lvl];
    const float* sc = scores + pixOfsA[lvl];

    for (int i = t; i < 4096; i += 1024) hist[i] = 0;
    __syncthreads();
    for (int i = t; i < n; i += 1024) {
        unsigned key = __float_as_uint(sc[i]) >> 20;   // scores in (0,1): monotone key
        atomicAdd(&hist[key], 1);
    }
    __syncthreads();

    if (t == 0) {
        int cum = 0, bsel = 0, need = 1000;
        for (int b = 4095; b >= 0; --b) {
            int c = hist[b];
            if (cum + c >= 1000) { bsel = b; need = 1000 - cum; break; }
            cum += c;
        }
        sh_thresh = bsel; sh_needEq = need; sh_runHi = 0; sh_runEq = 0;
    }
    __syncthreads();
    const int thresh = sh_thresh;
    const int needEq = sh_needEq;
    const int cntHi  = 1000 - needEq;

    // deterministic compaction in index order
    for (int base = 0; base < n; base += 1024) {
        int i = base + t;
        bool valid = (i < n);
        float s = valid ? sc[i] : 0.f;
        int key = valid ? (int)(__float_as_uint(s) >> 20) : -1;
        int isHi = (valid && key > thresh) ? 1 : 0;
        int isEq = (valid && key == thresh) ? 1 : 0;
        int totHi, totEq;
        int posHi = block_exscan(isHi, t, scanbuf, &totHi);
        int posEq = block_exscan(isEq, t, scanbuf, &totEq);
        if (isHi) {
            int slot = sh_runHi + posHi;           // [0, cntHi)
            selS[slot] = s; selI[slot] = i;
        }
        if (isEq) {
            int r = sh_runEq + posEq;
            if (r < needEq) { int slot = cntHi + r; selS[slot] = s; selI[slot] = i; }
        }
        __syncthreads();
        if (t == 0) { sh_runHi += totHi; sh_runEq += totEq; }
        __syncthreads();
    }
    if (t >= 1000) { selS[t] = -1.0f; selI[t] = 0x7fffffff; }
    __syncthreads();

    // bitonic sort 1024: score desc, index asc on ties (matches top_k stability)
    for (int k = 2; k <= 1024; k <<= 1) {
        for (int j = k >> 1; j > 0; j >>= 1) {
            int p = t ^ j;
            if (p > t) {
                float s0 = selS[t], s1 = selS[p];
                int i0 = selI[t], i1 = selI[p];
                bool before = (s0 > s1) || (s0 == s1 && i0 < i1);
                bool up = ((t & k) == 0);
                if (up ? !before : before) {
                    selS[t] = s1; selS[p] = s0;
                    selI[t] = i1; selI[p] = i0;
                }
            }
            __syncthreads();
        }
    }

    if (t < 1000) {
        int i = selI[t];
        if (i < n)
            decode_write(out, rowOfsA[lvl] + t, i, wA[lvl], strA[lvl], selS[t],
                         rg + (size_t)(pixOfsA[lvl] + i) * 4);
    }
}

// ---------- host launcher ----------
extern "C" void kernel_launch(void* const* d_in, const int* in_sizes, int n_in,
                              void* d_out, int out_size, void* d_ws, size_t ws_size,
                              hipStream_t stream) {
    (void)in_sizes; (void)n_in; (void)out_size; (void)ws_size;
    const float* feat[NLVL] = {(const float*)d_in[0], (const float*)d_in[1],
                               (const float*)d_in[2], (const float*)d_in[3],
                               (const float*)d_in[4]};
    const float* cls_w  = (const float*)d_in[5];
    const float* cls_b  = (const float*)d_in[6];
    const float* box_w  = (const float*)d_in[7];
    const float* box_b  = (const float*)d_in[8];
    const float* score_w = (const float*)d_in[9];
    const float* score_b = (const float*)d_in[10];
    const float* pred_w  = (const float*)d_in[11];
    const float* pred_b  = (const float*)d_in[12];
    const float* ctn_w   = (const float*)d_in[13];
    const float* ctn_b   = (const float*)d_in[14];

    const int Hs[NLVL] = {10, 20, 40, 80, 160};
    const int Ws[NLVL] = {15, 30, 60, 120, 240};
    const int HW[NLVL] = {150, 600, 2400, 9600, 38400};
    const int pixOfs[NLVL] = {0, 150, 750, 3150, 12750};        // unpadded (heads/scores)
    const int padOfs[NLVL] = {0, 204, 908, 3512, 13516};        // padded (H+2)*(W+2)
    const int rowOfs[NLVL] = {0, 150, 750, 1750, 2750};
    const float strideF[NLVL] = {128.f, 64.f, 32.f, 16.f, 8.f};

    // workspace carve (~93 MB total)
    uint8_t* ws = (uint8_t*)d_ws;
    size_t off = 0;
    auto carve = [&](size_t bytes) -> uint8_t* {
        uint8_t* p = ws + off;
        off = (off + bytes + 255) & ~(size_t)255;
        return p;
    };
    const size_t actBytes = (size_t)PADPIX * 256 * sizeof(h16);
    h16*   actIn  = (h16*)carve(actBytes);
    h16*   actA   = (h16*)carve(actBytes);
    h16*   actB   = (h16*)carve(actBytes);
    h16*   packCls = (h16*)carve((size_t)4 * 589824 * sizeof(h16));
    h16*   packBox = (h16*)carve((size_t)4 * 589824 * sizeof(h16));
    h16*   packHA  = (h16*)carve((size_t)36864 * sizeof(h16));
    h16*   packHB  = (h16*)carve((size_t)36864 * sizeof(h16));
    float* headBiasA = (float*)carve(16 * sizeof(float));
    float* headBiasB = (float*)carve(16 * sizeof(float));
    float* headsA = (float*)carve((size_t)TOTPIX * 16 * sizeof(float));
    float* headsB = (float*)carve((size_t)TOTPIX * 16 * sizeof(float));
    float* scores = (float*)carve((size_t)TOTPIX * sizeof(float));
    float* regs   = (float*)carve((size_t)TOTPIX * 4 * sizeof(float));
    float* out    = (float*)d_out;

    // 0) zero halos (whole act buffers; convs only ever write interiors)
    {
        int n = PADPIX * 256;
        zero_f16<<<(n + 255) / 256, 256, 0, stream>>>(actIn, n);
        zero_f16<<<(n + 255) / 256, 256, 0, stream>>>(actA, n);
        zero_f16<<<(n + 255) / 256, 256, 0, stream>>>(actB, n);
    }
    // 1) layout conversion into padded interiors
    for (int l = 0; l < NLVL; ++l) {
        int n = HW[l] * 256;
        to_nhwc_f16<<<(n + 255) / 256, 256, 0, stream>>>(
            feat[l], actIn + ((size_t)padOfs[l] << 8), Hs[l], Ws[l]);
    }
    // 2) weight packing
    for (int li = 0; li < 4; ++li) {
        pack_tower_w<<<589824 / 256, 256, 0, stream>>>(cls_w, packCls + (size_t)li * 589824, li);
        pack_tower_w<<<589824 / 256, 256, 0, stream>>>(box_w, packBox + (size_t)li * 589824, li);
    }
    pack_head_w<<<36864 / 256, 256, 0, stream>>>(score_w, nullptr, packHA, 0);
    pack_head_w<<<36864 / 256, 256, 0, stream>>>(pred_w, ctn_w, packHB, 1);
    build_head_bias<<<1, 16, 0, stream>>>(score_b, pred_b, ctn_b, headBiasA, headBiasB);

    // 3) towers + heads (cls then box; stream order preserves dependencies)
    for (int tower = 0; tower < 2; ++tower) {
        const h16*  wpkBase = tower ? packBox : packCls;
        const float* bias   = tower ? box_b : cls_b;
        for (int li = 0; li < 4; ++li) {
            const h16* in = (li == 0) ? actIn : ((li & 1) ? actA : actB);
            h16* ob = (li & 1) ? actB : actA;
            for (int l = 0; l < NLVL; ++l) {
                dim3 g((HW[l] + 63) / 64, 4);
                conv3x3_wmma<256, true><<<g, 128, 0, stream>>>(
                    in + ((size_t)padOfs[l] << 8), wpkBase + (size_t)li * 589824,
                    bias + li * 256, ob + ((size_t)padOfs[l] << 8), nullptr, Hs[l], Ws[l]);
            }
        }
        const h16* hw = tower ? packHB : packHA;
        const float* hb = tower ? headBiasB : headBiasA;
        float* hout = tower ? headsB : headsA;
        for (int l = 0; l < NLVL; ++l) {
            dim3 g((HW[l] + 127) / 128, 1);
            conv3x3_wmma<16, false><<<g, 128, 0, stream>>>(
                actB + ((size_t)padOfs[l] << 8), hw, hb, nullptr,
                hout + (size_t)pixOfs[l] * 16, Hs[l], Ws[l]);
        }
    }

    // 4) fused scores + reg gather
    fuse_scores<<<(TOTPIX + 255) / 256, 256, 0, stream>>>(headsA, headsB, scores, regs, TOTPIX);

    // 5) small levels: keep all anchors in index order
    for (int l = 0; l < 2; ++l) {
        emit_small<<<(HW[l] + 255) / 256, 256, 0, stream>>>(
            scores, regs, out, pixOfs[l], rowOfs[l], HW[l], Ws[l], strideF[l]);
    }
    // 6) big levels: top-1000 select + sort + decode
    topk_emit<<<3, 1024, 0, stream>>>(scores, regs, out);
}
